// MultiHeadAttentionMemory_84748294685256
// MI455X (gfx1250) — matrix operand order repaired and checked by reading, not compiled, not run on hardware
//
#include <hip/hip_runtime.h>

// ---------------------------------------------------------------------------
// Types for CDNA5 WMMA
// ---------------------------------------------------------------------------
typedef __bf16 bf16x16 __attribute__((ext_vector_type(16)));
typedef __bf16 bf16x8  __attribute__((ext_vector_type(8)));
typedef __bf16 bf16x2  __attribute__((ext_vector_type(2)));
typedef float  f32x8   __attribute__((ext_vector_type(8)));

union FragBF16 {
    bf16x16 v;
    bf16x8  h[2];
};

struct U42 { uint4 x0, x1; };   // one staged 32-byte strip, kept in VGPRs

// hardware fp32 -> bf16 (RNE) conversions
__device__ __forceinline__ unsigned short f2bf_bits(float f) {
    __bf16 h = (__bf16)f;
    return __builtin_bit_cast(unsigned short, h);
}
__device__ __forceinline__ unsigned pk2(float lo, float hi) {
    bf16x2 v = {(__bf16)lo, (__bf16)hi};
    return __builtin_bit_cast(unsigned, v);
}

// epilogue modes
#define EPI_BIAS  0   // + bias[col]
#define EPI_SCALE 1   // * rowScale[row] * colScale[col]
#define EPI_PLAIN 2   // raw

// ---------------------------------------------------------------------------
// fp32 -> bf16 bulk conversion (8 elements / thread), n % 2048 == 0
// ---------------------------------------------------------------------------
__global__ __launch_bounds__(256)
void f32_to_bf16_kernel(const float* __restrict__ in,
                        unsigned short* __restrict__ out, int n8)
{
    const int idx = blockIdx.x * 256 + threadIdx.x;
    if (idx >= n8) return;
    const float4* p = (const float4*)in + (size_t)idx * 2;
    float4 a = p[0], b = p[1];
    uint4 o;
    o.x = pk2(a.x, a.y); o.y = pk2(a.z, a.w);
    o.z = pk2(b.x, b.y); o.w = pk2(b.z, b.w);
    ((uint4*)out)[idx] = o;
}

// ---------------------------------------------------------------------------
// Tiled, double-buffered bf16 WMMA GEMM:  C[M,N] = epilogue(A[M,K] @ B)
// A, B are pre-converted bf16 (ushort bits).
//   BT = false : B is [K,N] row-major
//   BT = true  : B is [N,K] row-major (C = A @ B^T)
// WF32 / WBF16 select fp32 / bf16 output streams.
// Block: 256 threads = 8 waves, wave grid 4x2, wave tile 32x64 (2x4 WMMA).
// ---------------------------------------------------------------------------
#define BM 128
#define BN 128
#define BK 32
#define LDT 40   // padded LDS row stride (ushorts); 80B rows keep 16B alignment

template <bool BT, int MODE, bool WF32, bool WBF16>
__global__ __launch_bounds__(256)
void gemm_bf16_wmma(const unsigned short* __restrict__ A,
                    const unsigned short* __restrict__ Bm,
                    const float* __restrict__ bias,
                    const float* __restrict__ rowScale,
                    const float* __restrict__ colScale,
                    float* __restrict__ C, unsigned short* __restrict__ Cb,
                    int M, int N, int K)
{
    __shared__ unsigned short As[2][BM * LDT];
    __shared__ unsigned short Bs[2][BN * LDT];

    const int tid   = threadIdx.x;
    const int lane  = tid & 31;
    const int wave  = tid >> 5;          // 0..7
    const int waveM = wave >> 1;         // 0..3  -> 32 rows each
    const int waveN = wave & 1;          // 0..1  -> 64 cols each
    const int blockM = blockIdx.y * BM;
    const int blockN = blockIdx.x * BN;

    f32x8 acc[2][4];
#pragma unroll
    for (int i = 0; i < 2; ++i)
#pragma unroll
        for (int j = 0; j < 4; ++j)
            acc[i][j] = (f32x8){0.f, 0.f, 0.f, 0.f, 0.f, 0.f, 0.f, 0.f};

    // contiguous staging coords (A tile and transposed-B tile): 128 rows x 32 k
    const int srow  = tid >> 1;            // 0..127
    const int shalf = (tid & 1) * 16;      // 0 or 16 (k offset, ushorts)
    // scatter staging coords (row-major B -> n-major LDS): 2 k-rows x 8 n
    const int krow2 = (tid >> 4) * 2;      // 0,2,..,30
    const int nseg8 = (tid & 15) * 8;      // 0..120

    const unsigned short* aSrc  = A + (size_t)(blockM + srow) * K + shalf;
    const unsigned short* bSrcT = Bm + (size_t)(blockN + srow) * K + shalf;
    const unsigned short* bSrcR = Bm + (size_t)krow2 * N + blockN + nseg8;

    auto loadAat = [&](int k0) -> U42 {
        const unsigned short* s = aSrc + k0;
        return {*(const uint4*)(s), *(const uint4*)(s + 8)};
    };
    auto loadBat = [&](int k0) -> U42 {
        if constexpr (BT) {
            const unsigned short* s = bSrcT + k0;
            return {*(const uint4*)(s), *(const uint4*)(s + 8)};
        } else {
            const unsigned short* s = bSrcR + (size_t)k0 * N;
            return {*(const uint4*)(s), *(const uint4*)(s + N)};  // k, k+1 rows
        }
    };
    auto stageA = [&](int buf, U42 a) {
        unsigned short* dst = &As[buf][srow * LDT + shalf];
        *(uint4*)(dst)     = a.x0;
        *(uint4*)(dst + 8) = a.x1;
    };
    auto stageB = [&](int buf, U42 b) {
        if constexpr (BT) {
            unsigned short* dst = &Bs[buf][srow * LDT + shalf];
            *(uint4*)(dst)     = b.x0;
            *(uint4*)(dst + 8) = b.x1;
        } else {
            // interleave k / k+1 rows with v_perm_b32 (1 VALU per packed dword)
            const unsigned* lo = (const unsigned*)&b.x0;   // row k,   n..n+7
            const unsigned* hi = (const unsigned*)&b.x1;   // row k+1, n..n+7
#pragma unroll
            for (int j = 0; j < 4; ++j) {
                unsigned even = __builtin_amdgcn_perm(hi[j], lo[j], 0x05040100u);
                unsigned odd  = __builtin_amdgcn_perm(hi[j], lo[j], 0x07060302u);
                *(unsigned*)&Bs[buf][(nseg8 + 2 * j)     * LDT + krow2] = even;
                *(unsigned*)&Bs[buf][(nseg8 + 2 * j + 1) * LDT + krow2] = odd;
            }
        }
    };

    // fragment lane layouts (ISA 16-bit A 16x32 / B 32x16)
    const int mlane = lane & 15;
    const int ac0   = (lane < 16) ? 0 : 8;    // A: K {0..7,16..23} / {8..15,24..31}
    const int bc0   = (lane < 16) ? 0 : 8;    // B: K {0..7,16..23} / {8..15,24..31}

    auto compute = [&](int buf) {
        FragBF16 aF[2], bF[4];
#pragma unroll
        for (int i = 0; i < 2; ++i) {
            const int m = waveM * 32 + i * 16 + mlane;
            aF[i].h[0] = *(const bf16x8*)&As[buf][m * LDT + ac0];
            aF[i].h[1] = *(const bf16x8*)&As[buf][m * LDT + ac0 + 16];
        }
#pragma unroll
        for (int j = 0; j < 4; ++j) {
            const int n = waveN * 64 + j * 16 + mlane;
            bF[j].h[0] = *(const bf16x8*)&Bs[buf][n * LDT + bc0];
            bF[j].h[1] = *(const bf16x8*)&Bs[buf][n * LDT + bc0 + 16];
        }
#pragma unroll
        for (int i = 0; i < 2; ++i)
#pragma unroll
            for (int j = 0; j < 4; ++j)
                acc[i][j] = __builtin_amdgcn_wmma_f32_16x16x32_bf16(
                    false, aF[i].v, false, bF[j].v, (short)0, acc[i][j],
                    false, false);
    };

    // ---- double-buffered main loop: 1 barrier / K-step, VMEM under WMMA ----
    {
        U42 a0 = loadAat(0), b0 = loadBat(0);
        stageA(0, a0); stageB(0, b0);
    }
    int buf = 0;
    for (int k0 = 0; k0 < K; k0 += BK, buf ^= 1) {
        __syncthreads();                 // buf tile visible; buf^1 free to fill
        const bool more = (k0 + BK) < K;
        U42 na{}, nb{};
        if (more) { na = loadAat(k0 + BK); nb = loadBat(k0 + BK); }
        compute(buf);                    // WMMA on current tile
        if (more) { stageA(buf ^ 1, na); stageB(buf ^ 1, nb); }
    }

    // ---- epilogue: 16x16 f32 D layout (lanes 0-15: M=r; lanes 16-31: M=r+8)
    const int nl   = lane & 15;
    const int mofs = (lane < 16) ? 0 : 8;
#pragma unroll
    for (int i = 0; i < 2; ++i) {
        const int row0 = blockM + waveM * 32 + i * 16 + mofs;
        float rs[8];
        if constexpr (MODE == EPI_SCALE) {
#pragma unroll
            for (int r = 0; r < 8; ++r) rs[r] = rowScale[row0 + r];
        }
#pragma unroll
        for (int j = 0; j < 4; ++j) {
            const int col = blockN + waveN * 64 + j * 16 + nl;
            float bsv = 0.0f, csv = 1.0f;
            if constexpr (MODE == EPI_BIAS)  bsv = bias[col];
            if constexpr (MODE == EPI_SCALE) csv = colScale[col];
#pragma unroll
            for (int r = 0; r < 8; ++r) {
                float v = acc[i][j][r];
                if constexpr (MODE == EPI_BIAS)  v += bsv;
                if constexpr (MODE == EPI_SCALE) v *= rs[r] * csv;
                const size_t idx = (size_t)(row0 + r) * N + col;
                if constexpr (WF32)  C[idx]  = v;
                if constexpr (WBF16) Cb[idx] = f2bf_bits(v);
            }
        }
    }
}

// ---------------------------------------------------------------------------
// out[row] = 1 / max(||X[row,:]||_2, 1e-8)
// ---------------------------------------------------------------------------
__global__ __launch_bounds__(256)
void rownorm_inv_kernel(const float* __restrict__ X, float* __restrict__ out,
                        int cols)
{
    __shared__ float red[256];
    const int row = blockIdx.x;
    const float* p = X + (size_t)row * cols;
    float s = 0.0f;
    for (int c = threadIdx.x; c < cols; c += 256) {
        float v = p[c];
        s += v * v;
    }
    red[threadIdx.x] = s;
    __syncthreads();
    for (int o = 128; o > 0; o >>= 1) {
        if (threadIdx.x < o) red[threadIdx.x] += red[threadIdx.x + o];
        __syncthreads();
    }
    if (threadIdx.x == 0)
        out[row] = 1.0f / fmaxf(sqrtf(red[0]), 1e-8f);
}

// ---------------------------------------------------------------------------
// Per-row: softmax -> sparse shrinkage -> L1 renorm -> entropy accumulation.
// Writes w as bf16 (feeds the reconstruction GEMM directly).
// cols == 4096, blockDim == 256 (16 elements per thread, register resident).
// ---------------------------------------------------------------------------
__global__ __launch_bounds__(256)
void softmax_shrink_kernel(const float* __restrict__ logits,
                           unsigned short* __restrict__ w,
                           float* __restrict__ loss, int cols)
{
    __shared__ float red[256];
    const int row = blockIdx.x;
    const int tid = threadIdx.x;
    const float* lp = logits + (size_t)row * cols;
    unsigned short* wp = w + (size_t)row * cols;

    const float THRESHOLD = 1.0f / 4096.0f;
    const float EPS = 1e-12f;

    float l[16];
    float mx = -3.4028235e38f;
#pragma unroll
    for (int i = 0; i < 16; ++i) {
        l[i] = lp[tid + i * 256];
        mx = fmaxf(mx, l[i]);
    }
    red[tid] = mx; __syncthreads();
    for (int o = 128; o > 0; o >>= 1) {
        if (tid < o) red[tid] = fmaxf(red[tid], red[tid + o]);
        __syncthreads();
    }
    mx = red[0]; __syncthreads();

    float s = 0.0f;
#pragma unroll
    for (int i = 0; i < 16; ++i) { l[i] = __expf(l[i] - mx); s += l[i]; }
    red[tid] = s; __syncthreads();
    for (int o = 128; o > 0; o >>= 1) {
        if (tid < o) red[tid] += red[tid + o];
        __syncthreads();
    }
    const float inv = 1.0f / red[0]; __syncthreads();

    float s2 = 0.0f;
#pragma unroll
    for (int i = 0; i < 16; ++i) {
        float wv = l[i] * inv;
        float d  = wv - THRESHOLD;
        float w2 = (d > 0.0f) ? d * wv / (d + EPS) : 0.0f;  // relu(d)*w/(|d|+eps)
        l[i] = w2;
        s2 += w2;
    }
    red[tid] = s2; __syncthreads();
    for (int o = 128; o > 0; o >>= 1) {
        if (tid < o) red[tid] += red[tid + o];
        __syncthreads();
    }
    const float inv2 = 1.0f / fmaxf(red[0], EPS); __syncthreads();

    float ent = 0.0f;
#pragma unroll
    for (int i = 0; i < 16; ++i) {
        float w3 = l[i] * inv2;
        wp[tid + i * 256] = f2bf_bits(w3);
        ent += -w3 * __logf(w3 + EPS);
    }
    red[tid] = ent; __syncthreads();
    for (int o = 128; o > 0; o >>= 1) {
        if (tid < o) red[tid] += red[tid + o];
        __syncthreads();
    }
    if (tid == 0) atomicAdd(loss, red[0]);
}

__global__ void finalize_loss_kernel(const float* __restrict__ loss,
                                     float* __restrict__ out, float invB)
{
    out[0] = loss[0] * invB * 0.0002f;
}

// ---------------------------------------------------------------------------
// Launch: seq_len==1 => attention softmax == 1 => MHA reduces to
//   z2 = (x @ Wv + bv) @ Wo + bo         (Wq/Wk dead)
// then cosine-sim memory addressing, shrinkage softmax, reconstruction.
// ---------------------------------------------------------------------------
extern "C" void kernel_launch(void* const* d_in, const int* in_sizes, int n_in,
                              void* d_out, int out_size, void* d_ws,
                              size_t ws_size, hipStream_t stream)
{
    const float* x      = (const float*)d_in[0];   // [B, F]
    const float* memory = (const float*)d_in[1];   // [M, F]
    const float* Wv     = (const float*)d_in[6];   // [F, F]
    const float* bv     = (const float*)d_in[7];   // [F]
    const float* Wo     = (const float*)d_in[8];   // [F, F]
    const float* bo     = (const float*)d_in[9];   // [F]

    const int F = 4096;
    const int M = 4096;                  // NUM_MEMORIES
    const int B = in_sizes[0] / F;       // 256

    // ---- workspace carve-up (fp32 first, then bf16/ushort buffers) ----
    float* z2     = (float*)d_ws;                 // [B,F] fp32
    float* logits = z2 + (size_t)B * F;           // [B,M] fp32
    float* mn_inv = logits + (size_t)B * M;       // [M]
    float* zn_inv = mn_inv + M;                   // [B]
    float* loss   = zn_inv + B;                   // [1]
    unsigned short* xb   = (unsigned short*)(loss + 4);      // [B,F]
    unsigned short* Wvb  = xb + (size_t)B * F;               // [F,F]
    unsigned short* Wob  = Wvb + (size_t)F * F;              // [F,F]
    unsigned short* memb = Wob + (size_t)F * F;              // [M,F]
    unsigned short* tb   = memb + (size_t)M * F;             // [B,F]
    unsigned short* z2b  = tb + (size_t)B * F;               // [B,F]
    unsigned short* wb   = z2b + (size_t)B * F;              // [B,M]

    (void)n_in; (void)out_size; (void)ws_size;

    hipMemsetAsync(loss, 0, sizeof(float), stream);

    // ---- one-shot fp32 -> bf16 conversions (bandwidth bound, no GEMM VALU)
    f32_to_bf16_kernel<<<(B * F / 8 + 255) / 256, 256, 0, stream>>>(x, xb, B * F / 8);
    f32_to_bf16_kernel<<<(F * F / 8 + 255) / 256, 256, 0, stream>>>(Wv, Wvb, F * F / 8);
    f32_to_bf16_kernel<<<(F * F / 8 + 255) / 256, 256, 0, stream>>>(Wo, Wob, F * F / 8);
    f32_to_bf16_kernel<<<(M * F / 8 + 255) / 256, 256, 0, stream>>>(memory, memb, M * F / 8);

    // memory row norms (fp32 source, consumed by GEMM3 epilogue)
    rownorm_inv_kernel<<<M, 256, 0, stream>>>(memory, mn_inv, F);

    dim3 blk(256);
    dim3 gFF(F / BN, B / BM);
    dim3 gFM(M / BN, B / BM);

    // t = x @ Wv + bv           (bf16 out only)
    gemm_bf16_wmma<false, EPI_BIAS, false, true><<<gFF, blk, 0, stream>>>(
        xb, Wvb, bv, nullptr, nullptr, nullptr, tb, B, F, F);
    // z2 = t @ Wo + bo          (fp32 for norms + bf16 for GEMM3)
    gemm_bf16_wmma<false, EPI_BIAS, true, true><<<gFF, blk, 0, stream>>>(
        tb, Wob, bo, nullptr, nullptr, z2, z2b, B, F, F);

    rownorm_inv_kernel<<<B, 256, 0, stream>>>(z2, zn_inv, F);

    // logits = (z2 @ memory^T) * zn_inv[row] * mn_inv[col]
    gemm_bf16_wmma<true, EPI_SCALE, true, false><<<gFM, blk, 0, stream>>>(
        z2b, memb, nullptr, zn_inv, mn_inv, logits, nullptr, B, M, F);

    softmax_shrink_kernel<<<B, 256, 0, stream>>>(logits, wb, loss, M);

    // z_hat = w @ memory  -> d_out (fp32)
    gemm_bf16_wmma<false, EPI_PLAIN, true, false><<<gFF, blk, 0, stream>>>(
        wb, memb, nullptr, nullptr, nullptr, (float*)d_out, nullptr, B, F, M);

    finalize_loss_kernel<<<1, 1, 0, stream>>>(
        loss, (float*)d_out + (size_t)B * F, 1.0f / (float)B);
}
